// NNC_working_74887049773743
// MI455X (gfx1250) — matrix operand order repaired and checked by reading, not compiled, not run on hardware
//
#include <hip/hip_runtime.h>
#include <hip/hip_bf16.h>

// ---------------------------------------------------------------------------
// NNConv GNN restructured as a WMMA GEMM for gfx1250 (MI455X, wave32).
//
//   msg_e[o] = sum_{k,i} h_e[k] * x[src_e,i] * w2[k, i*32+o] + sum_i x_i*b2[i*32+o]
//            = (U @ W')[e, o],   U[e, k*3+i] = h_e[k]*x[src_e,i]  (K padded to 128,
//              rows 96..98 carry x for the b2 bias, 99..127 zero)
//
// Per 16-edge tile: 8x v_wmma_f32_16x16x32_f16 (K=128 in 4 chunks, N=32 in 2).
// ---------------------------------------------------------------------------

typedef _Float16 v16h __attribute__((ext_vector_type(16)));
typedef float    v8f  __attribute__((ext_vector_type(8)));

#define NNODES   50000
#define NEDGES   1600000
#define EMBD     32
#define NTILES   (NEDGES / 16)      // 100000 exactly

// workspace layout (bytes)
#define AGG_BYTES   (NNODES * EMBD * 4)        // 6,400,000  (16B aligned)
#define BP_OFF      AGG_BYTES                  // 128x32 f16 = 8192 B
#define EMB_OFF     (BP_OFF + 128 * 32 * 2)    // 16x32 f32 bits (as int)

union FragH { v16h v; uint4 q[2]; };

static __device__ __forceinline__ unsigned pkh(float a, float b) {
  union { _Float16 h[2]; unsigned u; } x;
  x.h[0] = (_Float16)a; x.h[1] = (_Float16)b; return x.u;
}

// --- zero helper -----------------------------------------------------------
__global__ void nnc_zero_f32(float* __restrict__ p, int n) {
  int i = blockIdx.x * blockDim.x + threadIdx.x;
  if (i < n) p[i] = 0.f;
}

// --- prep: build W' (K=128, N=32) in f16, column-major [n][K] --------------
__global__ void nnc_prep(const float* __restrict__ w2, const float* __restrict__ b2,
                         _Float16* __restrict__ bp) {
  for (int idx = threadIdx.x; idx < 32 * 128; idx += blockDim.x) {
    int n = idx >> 7;        // column (output channel)
    int K = idx & 127;       // padded K index
    float v = 0.f;
    if (K < 96)      { int k = K / 3, i = K - 3 * k; v = w2[k * 96 + i * 32 + n]; }
    else if (K < 99) { v = b2[(K - 96) * 32 + n]; }
    bp[idx] = (_Float16)v;
  }
}

// --- main edge GEMM + scatter ----------------------------------------------
__global__ __launch_bounds__(256)
void nnc_edge_kernel(const float* __restrict__ x,
                     const float* __restrict__ edge_attr,
                     const float* __restrict__ w1,
                     const float* __restrict__ b1,
                     const int*   __restrict__ eidx,   // [2, E]: src row, dst row
                     const _Float16* __restrict__ bp,  // W' fragments source
                     float* __restrict__ agg) {
  // per-wave private U tile: 16 rows x 136 halfs (272B pitch, bank-friendly)
  __shared__ __align__(16) char smem[8 * 16 * 272];

  const int lane = threadIdx.x & 31;
  const int wave = threadIdx.x >> 5;
  const int n16  = lane & 15;     // row (A: M=lane%16; B/C column = lane%16)
  const int g    = lane >> 4;     // half-wave group
  const int gw   = blockIdx.x * 8 + wave;
  const int nw   = gridDim.x * 8;

  // --- preload B fragments (constant W') into VGPRs: 4 K-chunks x 2 N-chunks
  // B 32x16 f16, this code's k order (same as A): lane holds column n, halfs K=[kb+16g, kb+16g+16) contiguous.
  FragH B[4][2];
#pragma unroll
  for (int kc = 0; kc < 4; ++kc)
#pragma unroll
    for (int nc = 0; nc < 2; ++nc) {
      const uint4* p = (const uint4*)(bp + ((nc * 16 + n16) * 128 + kc * 32 + 16 * g));
      B[kc][nc].q[0] = p[0];
      B[kc][nc].q[1] = p[1];
    }

  // --- per-lane slice of the edge MLP weights: k in [16g, 16g+16)
  float w1r[16], b1r[16];
  {
    const float4* wp = (const float4*)w1;
    const float4* bb = (const float4*)b1;
#pragma unroll
    for (int q = 0; q < 4; ++q) {
      float4 t = wp[g * 4 + q];
      w1r[4 * q] = t.x; w1r[4 * q + 1] = t.y; w1r[4 * q + 2] = t.z; w1r[4 * q + 3] = t.w;
      float4 u = bb[g * 4 + q];
      b1r[4 * q] = u.x; b1r[4 * q + 1] = u.y; b1r[4 * q + 2] = u.z; b1r[4 * q + 3] = u.w;
    }
  }

  char* myrow = smem + wave * (16 * 272) + n16 * 272;
  // zero halfs K=[96,104) and [112,120) of my row once (tail stays zero across tiles)
  *(uint4*)(myrow + 192 + 32 * g) = make_uint4(0u, 0u, 0u, 0u);

  for (int tile = gw; tile < NTILES; tile += nw) {
    const int e = tile * 16 + n16;            // lanes L and L+16 handle the same edge
    const float a = edge_attr[e];
    const int   s = eidx[e];                  // src node
    const int   d = eidx[NEDGES + e];         // dst node
    const float x0 = x[3 * s], x1 = x[3 * s + 1], x2 = x[3 * s + 2];

    // build my 48-half slice of U row (K in [48g, 48g+48)): u[3k+i] = h[k]*x[i]
    unsigned pk[24];
#pragma unroll
    for (int p = 0; p < 8; ++p) {
      float h0 = fmaxf(fmaf(a, w1r[2 * p],     b1r[2 * p]),     0.f);
      float h1 = fmaxf(fmaf(a, w1r[2 * p + 1], b1r[2 * p + 1]), 0.f);
      pk[3 * p + 0] = pkh(h0 * x0, h0 * x1);
      pk[3 * p + 1] = pkh(h0 * x2, h1 * x0);
      pk[3 * p + 2] = pkh(h1 * x1, h1 * x2);
    }
    uint4* dp = (uint4*)(myrow + 96 * g);
#pragma unroll
    for (int t = 0; t < 6; ++t)
      dp[t] = make_uint4(pk[4 * t], pk[4 * t + 1], pk[4 * t + 2], pk[4 * t + 3]);
    // bias rows K=96..98 carry x (pads 99..103 rewritten as zero)
    if (g == 0)
      *(uint4*)(myrow + 192) = make_uint4(pkh(x0, x1), pkh(x2, 0.f), 0u, 0u);

    // LDS store->load fence (wave-private region; also a compiler barrier)
    asm volatile("s_wait_dscnt 0" ::: "memory");

    v8f c0 = {0.f, 0.f, 0.f, 0.f, 0.f, 0.f, 0.f, 0.f};
    v8f c1 = {0.f, 0.f, 0.f, 0.f, 0.f, 0.f, 0.f, 0.f};
#pragma unroll
    for (int kc = 0; kc < 4; ++kc) {
      // A 16x32 f16, this code's k order (same as B): lane row = lane%16; halfs K=[kb+16g, kb+16g+16) contiguous
      FragH A;
      A.q[0] = *(const uint4*)(myrow + kc * 64 + 32 * g);
      A.q[1] = *(const uint4*)(myrow + kc * 64 + 32 * g + 16);
      c0 = __builtin_amdgcn_wmma_f32_16x16x32_f16(false, A.v, false, B[kc][0].v,
                                                  (short)0, c0, false, false);
      c1 = __builtin_amdgcn_wmma_f32_16x16x32_f16(false, A.v, false, B[kc][1].v,
                                                  (short)0, c1, false, false);
    }

    // scatter-add: C layout lane holds N=lane%16, VGPR r holds M=(lane/16)*8+r
    float* aggc = agg + n16;
#pragma unroll
    for (int r = 0; r < 8; ++r) {
      int m  = (g << 3) + r;
      int dr = __shfl(d, m, 32);      // dst of edge row m (held by lane m)
      atomicAdd(aggc + dr * EMBD,      c0[r]);
      atomicAdd(aggc + dr * EMBD + 16, c1[r]);
    }
  }
}

// --- node update: h = relu(agg + x@root + bias); segment max via int atomicMax
// (valid because h >= 0 and emb buffer initialized to 0 bits == 0.0f)
__global__ void nnc_node_kernel(const float* __restrict__ x,
                                const float* __restrict__ root,
                                const float* __restrict__ conv_bias,
                                const int*   __restrict__ batch,
                                const float* __restrict__ agg,
                                int* __restrict__ embi) {
  int gid = blockIdx.x * blockDim.x + threadIdx.x;
  if (gid >= NNODES * EMBD) return;
  int n = gid >> 5, o = gid & 31;
  float v = agg[gid] + conv_bias[o];
  v = fmaf(x[3 * n],     root[o],      v);
  v = fmaf(x[3 * n + 1], root[32 + o], v);
  v = fmaf(x[3 * n + 2], root[64 + o], v);
  v = fmaxf(v, 0.f);
  atomicMax(embi + batch[n] * EMBD + o, __float_as_int(v));
}

// --- final FC: (16 x 32) @ (32 x 2) -----------------------------------------
__global__ void nnc_fc_kernel(const int* __restrict__ embi,
                              const float* __restrict__ fc_w,
                              const float* __restrict__ fc_b,
                              float* __restrict__ out) {
  int t = threadIdx.x;
  if (t >= 32) return;
  int gph = t >> 1, c = t & 1;
  float s = fc_b[c];
#pragma unroll
  for (int o = 0; o < 32; ++o) {
    float e = fmaxf(__int_as_float(embi[gph * 32 + o]), 0.f);
    s = fmaf(e, fc_w[o * 2 + c], s);
  }
  out[gph * 2 + c] = s;
}

extern "C" void kernel_launch(void* const* d_in, const int* in_sizes, int n_in,
                              void* d_out, int out_size, void* d_ws, size_t ws_size,
                              hipStream_t stream) {
  const float* x         = (const float*)d_in[0];
  const float* edge_attr = (const float*)d_in[1];
  const float* w1        = (const float*)d_in[2];
  const float* b1        = (const float*)d_in[3];
  const float* w2        = (const float*)d_in[4];
  const float* b2        = (const float*)d_in[5];
  const float* root      = (const float*)d_in[6];
  const float* conv_bias = (const float*)d_in[7];
  const float* fc_w      = (const float*)d_in[8];
  const float* fc_b      = (const float*)d_in[9];
  const int*   eidx      = (const int*)d_in[10];
  const int*   batch     = (const int*)d_in[11];
  float* out = (float*)d_out;

  char* ws = (char*)d_ws;
  float*     agg  = (float*)ws;
  _Float16*  bp   = (_Float16*)(ws + BP_OFF);
  int*       embi = (int*)(ws + EMB_OFF);

  nnc_zero_f32<<<(NNODES * EMBD + 255) / 256, 256, 0, stream>>>(agg, NNODES * EMBD);
  nnc_zero_f32<<<2, 256, 0, stream>>>((float*)embi, 16 * EMBD);
  nnc_prep<<<1, 256, 0, stream>>>(w2, b2, bp);
  nnc_edge_kernel<<<2000, 256, 0, stream>>>(x, edge_attr, w1, b1, eidx, bp, agg);
  nnc_node_kernel<<<(NNODES * EMBD + 255) / 256, 256, 0, stream>>>(x, root, conv_bias,
                                                                   batch, agg, embi);
  nnc_fc_kernel<<<1, 32, 0, stream>>>(embi, fc_w, fc_b, out);
}
